// CausalSelfAttention_40192303956187
// MI455X (gfx1250) — matrix operand order repaired and checked by reading, not compiled, not run on hardware
//
#include <hip/hip_runtime.h>

typedef __attribute__((ext_vector_type(16))) _Float16 v16h;
typedef __attribute__((ext_vector_type(8)))  _Float16 v8h;
typedef __attribute__((ext_vector_type(8)))  float    v8f;

#define BB 2
#define TT 4096
#define DD 512
#define HH 8
#define HD 64
#define MM (BB * TT)   // 8192 rows

union V16U { v16h v; v8h h[2]; };

// A fragment (16x32 f16): lane l holds row m=l&15,
// halves h = K (hi*16 + h), matching the B fragment order
__device__ __forceinline__ v16h load_a_frag(const _Float16* base, int ld, int row0, int k0, int lane) {
    int m = lane & 15, hi = lane >> 4;
    const _Float16* p = base + (size_t)(row0 + m) * ld + k0 + hi * 16;
    V16U u;
    u.h[0] = *(const v8h*)(p);
    u.h[1] = *(const v8h*)(p + 8);
    return u.v;
}

// B fragment (32x16 f16) from a [N][K] row-major array: lane l holds col n=l&15,
// halves h = K (hi*16 + h): one contiguous 32-byte load.
__device__ __forceinline__ v16h load_b_frag(const _Float16* base, int ld, int n0, int k0, int lane) {
    int n = lane & 15, hi = lane >> 4;
    return *(const v16h*)(base + (size_t)(n0 + n) * ld + k0 + hi * 16);
}

__global__ void cvt_f32_to_f16_40192303956187(const float* __restrict__ in, _Float16* __restrict__ out, int n) {
    int i = blockIdx.x * blockDim.x + threadIdx.x;
    if (i < n) out[i] = (_Float16)in[i];
}

// qkv = x @ w_qkv^T ; each wave: 32x64 block (2 A-frags x 4 B-frags -> 8 WMMAs/K-step,
// every B fragment reused across both M rows: 12 b128 loads per 8 WMMAs).
// scatter into Q [bh][t][d], K [bh][t][d], Vt [bh][d][t]
__global__ void qkv_gemm_40192303956187(const _Float16* __restrict__ xh, const _Float16* __restrict__ wq,
                                        _Float16* __restrict__ qh, _Float16* __restrict__ kh,
                                        _Float16* __restrict__ vt) {
    int lane = threadIdx.x & 31;
    int w    = threadIdx.x >> 5;
    int row0 = blockIdx.x * 32;                   // 32-row M block
    int n0   = (blockIdx.y * 4 + w) * 64;         // 64-wide N strip (N = 1536 -> 24 strips)
    v8f acc[2][4];
#pragma unroll
    for (int mi = 0; mi < 2; ++mi)
#pragma unroll
        for (int j = 0; j < 4; ++j) acc[mi][j] = v8f{};
#pragma unroll
    for (int k = 0; k < DD; k += 32) {
        v16h a0 = load_a_frag(xh, DD, row0,      k, lane);
        v16h a1 = load_a_frag(xh, DD, row0 + 16, k, lane);
#pragma unroll
        for (int j = 0; j < 4; ++j) {
            v16h b = load_b_frag(wq, DD, n0 + j * 16, k, lane);
            acc[0][j] = __builtin_amdgcn_wmma_f32_16x16x32_f16(false, a0, false, b, (short)0, acc[0][j], false, false);
            acc[1][j] = __builtin_amdgcn_wmma_f32_16x16x32_f16(false, a1, false, b, (short)0, acc[1][j], false, false);
        }
    }
    int hi  = lane >> 4;
    int nl  = lane & 15;
#pragma unroll
    for (int j = 0; j < 4; ++j) {
        int n   = n0 + j * 16 + nl;
        int sec = n / DD;                 // 0=Q 1=K 2=V  (uniform per strip)
        int nn  = n % DD;
        int h   = nn / HD, d = nn % HD;   // strip spans exactly one head
#pragma unroll
        for (int mi = 0; mi < 2; ++mi)
#pragma unroll
            for (int r = 0; r < 8; ++r) {
                int mg = row0 + mi * 16 + r + hi * 8;
                int b_ = mg / TT, t = mg % TT;
                size_t bh = (size_t)b_ * HH + h;
                _Float16 val = (_Float16)acc[mi][j][r];
                if (sec == 0)      qh[(bh * TT + t) * HD + d] = val;
                else if (sec == 1) kh[(bh * TT + t) * HD + d] = val;
                else               vt[(bh * HD + d) * TT + t] = val;   // pre-transposed V
            }
    }
}

// Flash-attention: one wave per 32-query tile (two 16-row subtiles sharing K/V frags),
// online softmax in f32, f16 WMMA everywhere.
__global__ void attn_40192303956187(const _Float16* __restrict__ qh, const _Float16* __restrict__ kh,
                                    const _Float16* __restrict__ vt, _Float16* __restrict__ ao) {
    __shared__ __align__(32) _Float16 plds[4][2][16 * 48];   // per-wave, per-subtile P tiles
    int lane = threadIdx.x & 31;
    int w    = threadIdx.x >> 5;
    int hi   = lane >> 4;
    int nl   = lane & 15;
    int bh   = blockIdx.y;
    int qb0  = (blockIdx.x * 4 + w) * 32;        // 32-aligned query base
    const _Float16* Q = qh + (size_t)bh * TT * HD;
    const _Float16* K = kh + (size_t)bh * TT * HD;
    const _Float16* V = vt + (size_t)bh * HD * TT;

    v16h qa[2][2];
#pragma unroll
    for (int qs = 0; qs < 2; ++qs) {
        qa[qs][0] = load_a_frag(Q, HD, qb0 + qs * 16, 0, lane);
        qa[qs][1] = load_a_frag(Q, HD, qb0 + qs * 16, 32, lane);
    }
    v8f o[2][4];
    float rmax[2][8], rsum[2][8];
#pragma unroll
    for (int qs = 0; qs < 2; ++qs) {
#pragma unroll
        for (int r = 0; r < 8; ++r) { rmax[qs][r] = -1e30f; rsum[qs][r] = 0.0f; }
#pragma unroll
        for (int j = 0; j < 4; ++j) o[qs][j] = v8f{};
    }

    int lastkb = qb0;                            // rows qb0..qb0+31 -> last chunk = qb0
    for (int kb = 0; kb <= lastkb; kb += 32) {
        bool maskchunk = (kb == lastkb);
        // K-tile B fragments, shared by both query subtiles
        v16h kf0[2], kf1[2];
#pragma unroll
        for (int ns = 0; ns < 2; ++ns) {
            kf0[ns] = load_b_frag(K, HD, kb + ns * 16, 0, lane);
            kf1[ns] = load_b_frag(K, HD, kb + ns * 16, 32, lane);
        }
#pragma unroll
        for (int qs = 0; qs < 2; ++qs) {
            v8f s[2];
#pragma unroll
            for (int ns = 0; ns < 2; ++ns) {
                v8f c = {};
                c = __builtin_amdgcn_wmma_f32_16x16x32_f16(false, qa[qs][0], false, kf0[ns], (short)0, c, false, false);
                c = __builtin_amdgcn_wmma_f32_16x16x32_f16(false, qa[qs][1], false, kf1[ns], (short)0, c, false, false);
                s[ns] = c;
            }
            float cmax[8];
#pragma unroll
            for (int r = 0; r < 8; ++r) {
                int q = qb0 + qs * 16 + r + hi * 8;
                float v0 = (float)s[0][r] * 0.125f;   // 1/sqrt(64)
                float v1 = (float)s[1][r] * 0.125f;
                if (maskchunk) {
                    if (kb + nl      > q) v0 = -1e30f;
                    if (kb + 16 + nl > q) v1 = -1e30f;
                }
                s[0][r] = v0; s[1][r] = v1;
                float mx = fmaxf(v0, v1);
#pragma unroll
                for (int off = 1; off < 16; off <<= 1)
                    mx = fmaxf(mx, __shfl_xor(mx, off, 32));
                cmax[r] = mx;
            }
            float corr[8];
#pragma unroll
            for (int r = 0; r < 8; ++r) {
                float nm = fmaxf(rmax[qs][r], cmax[r]);
                corr[r] = __expf(rmax[qs][r] - nm);
                rmax[qs][r] = nm;
            }
            _Float16* myp = &plds[w][qs][0];
#pragma unroll
            for (int r = 0; r < 8; ++r) {
                float p0 = __expf((float)s[0][r] - rmax[qs][r]);
                float p1 = __expf((float)s[1][r] - rmax[qs][r]);
                int m = r + hi * 8;
                myp[m * 48 + nl]      = (_Float16)p0;
                myp[m * 48 + 16 + nl] = (_Float16)p1;
                float sm = p0 + p1;
#pragma unroll
                for (int off = 1; off < 16; off <<= 1)
                    sm += __shfl_xor(sm, off, 32);
                rsum[qs][r] = rsum[qs][r] * corr[r] + sm;
            }
#pragma unroll
            for (int r = 0; r < 8; ++r) {
                o[qs][0][r] *= corr[r]; o[qs][1][r] *= corr[r];
                o[qs][2][r] *= corr[r]; o[qs][3][r] *= corr[r];
            }
        }
        // O += P @ V : V-tile B fragments loaded once, used by both subtiles
        v16h pa0 = load_a_frag(&plds[w][0][0], 48, 0, 0, lane);
        v16h pa1 = load_a_frag(&plds[w][1][0], 48, 0, 0, lane);
#pragma unroll
        for (int j = 0; j < 4; ++j) {
            v16h vb = load_b_frag(V, TT, j * 16, kb, lane);
            o[0][j] = __builtin_amdgcn_wmma_f32_16x16x32_f16(false, pa0, false, vb, (short)0, o[0][j], false, false);
            o[1][j] = __builtin_amdgcn_wmma_f32_16x16x32_f16(false, pa1, false, vb, (short)0, o[1][j], false, false);
        }
    }
    // normalize, merge heads: ao[b][t][h*64 + d]
    int b_ = bh / HH, h = bh % HH;
#pragma unroll
    for (int qs = 0; qs < 2; ++qs)
#pragma unroll
        for (int r = 0; r < 8; ++r) {
            int t = qb0 + qs * 16 + r + hi * 8;
            size_t rowoff = ((size_t)b_ * TT + t) * DD + (size_t)h * HD;
            float inv = 1.0f / rsum[qs][r];
            ao[rowoff +  0 + nl] = (_Float16)(o[qs][0][r] * inv);
            ao[rowoff + 16 + nl] = (_Float16)(o[qs][1][r] * inv);
            ao[rowoff + 32 + nl] = (_Float16)(o[qs][2][r] * inv);
            ao[rowoff + 48 + nl] = (_Float16)(o[qs][3][r] * inv);
        }
}

// out = attn_out @ w_out^T  (f32 output), 32x64 block per wave
__global__ void out_gemm_40192303956187(const _Float16* __restrict__ ah, const _Float16* __restrict__ wo,
                                        float* __restrict__ out) {
    int lane = threadIdx.x & 31;
    int w    = threadIdx.x >> 5;
    int row0 = blockIdx.x * 32;
    int n0   = (blockIdx.y * 4 + w) * 64;        // N = 512 -> 8 strips
    v8f acc[2][4];
#pragma unroll
    for (int mi = 0; mi < 2; ++mi)
#pragma unroll
        for (int j = 0; j < 4; ++j) acc[mi][j] = v8f{};
#pragma unroll
    for (int k = 0; k < DD; k += 32) {
        v16h a0 = load_a_frag(ah, DD, row0,      k, lane);
        v16h a1 = load_a_frag(ah, DD, row0 + 16, k, lane);
#pragma unroll
        for (int j = 0; j < 4; ++j) {
            v16h b = load_b_frag(wo, DD, n0 + j * 16, k, lane);
            acc[0][j] = __builtin_amdgcn_wmma_f32_16x16x32_f16(false, a0, false, b, (short)0, acc[0][j], false, false);
            acc[1][j] = __builtin_amdgcn_wmma_f32_16x16x32_f16(false, a1, false, b, (short)0, acc[1][j], false, false);
        }
    }
    int hi = lane >> 4;
    int nl = lane & 15;
#pragma unroll
    for (int j = 0; j < 4; ++j) {
        int n = n0 + j * 16 + nl;
#pragma unroll
        for (int mi = 0; mi < 2; ++mi)
#pragma unroll
            for (int r = 0; r < 8; ++r) {
                int mg = row0 + mi * 16 + r + hi * 8;
                out[(size_t)mg * DD + n] = acc[mi][j][r];
            }
    }
}

extern "C" void kernel_launch(void* const* d_in, const int* in_sizes, int n_in,
                              void* d_out, int out_size, void* d_ws, size_t ws_size,
                              hipStream_t stream) {
    const float* x    = (const float*)d_in[0];
    const float* wqkv = (const float*)d_in[1];
    const float* wout = (const float*)d_in[2];
    float* out = (float*)d_out;

    _Float16* ws = (_Float16*)d_ws;
    size_t off = 0;
    _Float16* xh   = ws + off; off += (size_t)MM * DD;        // x in f16
    _Float16* wq16 = ws + off; off += (size_t)3 * DD * DD;    // w_qkv f16 [N][K]
    _Float16* wo16 = ws + off; off += (size_t)DD * DD;        // w_out f16 [N][K]
    _Float16* qh   = ws + off; off += (size_t)MM * DD;        // Q  [bh][t][d]
    _Float16* kh   = ws + off; off += (size_t)MM * DD;        // K  [bh][t][d]
    _Float16* vt   = ws + off; off += (size_t)MM * DD;        // Vt [bh][d][t]
    _Float16* ao   = ws + off; off += (size_t)MM * DD;        // attn out [m][d]

    int nx = MM * DD, nq = 3 * DD * DD, no = DD * DD;
    cvt_f32_to_f16_40192303956187<<<(nx + 255) / 256, 256, 0, stream>>>(x, xh, nx);
    cvt_f32_to_f16_40192303956187<<<(nq + 255) / 256, 256, 0, stream>>>(wqkv, wq16, nq);
    cvt_f32_to_f16_40192303956187<<<(no + 255) / 256, 256, 0, stream>>>(wout, wo16, no);

    // M=8192 -> 256 row blocks of 32; N=1536 -> 24 strips of 64 (4 waves/block)
    qkv_gemm_40192303956187<<<dim3(MM / 32, 6), 128, 0, stream>>>(xh, wq16, qh, kh, vt);
    // T/(4 waves * 32 queries) x (B*H)
    attn_40192303956187<<<dim3(TT / 128, BB * HH), 128, 0, stream>>>(qh, kh, vt, ao);
    // N=512 -> 8 strips of 64
    out_gemm_40192303956187<<<dim3(MM / 32, 2), 128, 0, stream>>>(ao, wo16, out);
}
